// LaMO_88751204204627
// MI455X (gfx1250) — matrix operand order repaired and checked by reading, not compiled, not run on hardware
//
#include <hip/hip_runtime.h>
#include <hip/hip_bf16.h>
#include <math.h>

#define BQ 2
#define LL 4096
#define DM 192
#define DI 384
#define HH 48
#define NC 64

typedef __attribute__((ext_vector_type(16))) __bf16 bf16x16;
typedef __attribute__((ext_vector_type(8)))  float  f32x8;
typedef __attribute__((ext_vector_type(4)))  unsigned int u32x4;

union Frag { bf16x16 v; u32x4 u[2]; };

__device__ __forceinline__ f32x8 wmma_bf16(const Frag& a, const Frag& b, f32x8 c) {
  return __builtin_amdgcn_wmma_f32_16x16x32_bf16(false, a.v, false, b.v, (short)0, c, false, false);
}
__device__ __forceinline__ void zero8(f32x8& a){
#pragma unroll
  for (int i=0;i<8;++i) a[i]=0.0f;
}
// A-frag: 16x32 bf16, row-major source [M][K]; per ISA 7.12.2 (16-bit A layout)
__device__ __forceinline__ void load_a(Frag& f, const __bf16* base, int ld, int row0, int kb, int lane){
  const __bf16* p = base + (size_t)(row0 + (lane&15))*ld + kb + ((lane>>4)<<3);
  f.u[0] = *(const u32x4*)(p);
  f.u[1] = *(const u32x4*)(p+16);
}
// B-frag: 32x16 bf16; source stored [N][K] row-major (K contiguous per column)
__device__ __forceinline__ void load_b(Frag& f, const __bf16* base, int ld, int col0, int kb, int lane){
  const __bf16* p = base + (size_t)(col0 + (lane&15))*ld + kb + ((lane>>4)<<3);
  f.u[0] = *(const u32x4*)(p);
  f.u[1] = *(const u32x4*)(p+16);
}
__device__ __forceinline__ float gelu_f(float x){ return 0.5f*x*(1.0f+erff(x*0.70710678118f)); }
__device__ __forceinline__ float softplus_f(float x){ return x>20.f ? x : log1pf(expf(x)); }
__device__ __forceinline__ int dirmap(int d, int l){
  switch(d){
    case 1: return ((l&63)<<6)|(l>>6);
    case 2: return 4095-l;
    case 3: { int t=4095-l; return ((t&63)<<6)|(t>>6); }
    default: return l;
  }
}

// ---------- f32 -> bf16 conversion ----------
__global__ void k_cvt_bf16(const float* __restrict__ in, __bf16* __restrict__ out, int n){
  int i = blockIdx.x*blockDim.x+threadIdx.x;
  if (i<n) out[i] = (__bf16)in[i];
}

// ---------- in_proj GEMM: [8192x192] x [768x192]^T, split + gelu(z) ----------
__global__ void __launch_bounds__(256) k_inproj(const __bf16* __restrict__ xbf, const __bf16* __restrict__ wbf,
                                                float* __restrict__ xi_pre, float* __restrict__ zg){
  int lane = threadIdx.x & 31;
  int gw = blockIdx.x*8 + (threadIdx.x>>5);
  int mt = gw/12, ng = gw%12;
  int row0 = mt*16, col0 = ng*64;
  f32x8 acc[4];
#pragma unroll
  for (int t=0;t<4;++t) zero8(acc[t]);
#pragma unroll
  for (int kb=0;kb<DM;kb+=32){
    Frag a; load_a(a, xbf, DM, row0, kb, lane);
#pragma unroll
    for (int t=0;t<4;++t){ Frag b; load_b(b, wbf, DM, col0+t*16, kb, lane); acc[t]=wmma_bf16(a,b,acc[t]); }
  }
  int rbase = row0 + ((lane>>4)<<3);
#pragma unroll
  for (int t=0;t<4;++t){
    int col = col0 + t*16 + (lane&15);
#pragma unroll
    for (int j=0;j<8;++j){
      int row = rbase + j;
      float v = acc[t][j];
      if (col < DI) xi_pre[(size_t)row*DI + col] = v;
      else          zg[(size_t)row*DI + col - DI] = gelu_f(v);
    }
  }
}

// ---------- depthwise 3x3 conv + bias + gelu ----------
__global__ void k_conv(const float* __restrict__ xi_pre, const float* __restrict__ cw,
                       const float* __restrict__ cb, float* __restrict__ xconv, __bf16* __restrict__ xcbf){
  int idx = blockIdx.x*blockDim.x + threadIdx.x;
  if (idx >= BQ*LL*DI) return;
  int d = idx % DI; int t = idx / DI; int l = t & 4095; int b = t >> 12;
  int h = l >> 6, w = l & 63;
  float s = cb[d];
#pragma unroll
  for (int kh=0;kh<3;++kh){
    int hh = h + kh - 1;
    if ((unsigned)hh >= 64u) continue;
#pragma unroll
    for (int kw=0;kw<3;++kw){
      int ww = w + kw - 1;
      if ((unsigned)ww >= 64u) continue;
      s += xi_pre[((size_t)(b*LL) + (size_t)(hh*64+ww))*DI + d] * cw[d*9 + kh*3 + kw];
    }
  }
  s = gelu_f(s);
  xconv[idx] = s;
  xcbf[idx]  = (__bf16)s;
}

// ---------- x_proj GEMM (4 directions, mapped A rows, N=140) ----------
__global__ void __launch_bounds__(256) k_xproj(const __bf16* __restrict__ xcbf, const __bf16* __restrict__ wxp,
                                               float* __restrict__ dts, float* __restrict__ Bs, float* __restrict__ Cs){
  int lane = threadIdx.x & 31;
  int gw = blockIdx.x*8 + (threadIdx.x>>5);
  int ng = gw % 3; int tmp = gw/3; int mt = tmp & 511; int dir = tmp >> 9;
  int row0 = mt*16;
  int g = row0 + (lane&15);
  int b = g >> 12, l = g & 4095;
  int lm = dirmap(dir, l);
  const __bf16* abase = xcbf + ((size_t)(b*LL + lm))*DI;
  const __bf16* wb    = wxp + (size_t)dir*140*DI;
  f32x8 acc[3];
#pragma unroll
  for (int t=0;t<3;++t) zero8(acc[t]);
#pragma unroll
  for (int kb=0;kb<DI;kb+=32){
    Frag a;
    { const __bf16* p = abase + kb + ((lane>>4)<<3);
      a.u[0]=*(const u32x4*)p; a.u[1]=*(const u32x4*)(p+16); }
#pragma unroll
    for (int t=0;t<3;++t){
      int col = ng*48 + t*16 + (lane&15);
      Frag bf2;
      if (col < 140){
        const __bf16* p = wb + (size_t)col*DI + kb + ((lane>>4)<<3);
        bf2.u[0]=*(const u32x4*)p; bf2.u[1]=*(const u32x4*)(p+16);
      } else {
#pragma unroll
        for (int i=0;i<4;++i){ bf2.u[0][i]=0u; bf2.u[1][i]=0u; }
      }
      acc[t]=wmma_bf16(a,bf2,acc[t]);
    }
  }
  int rbase = row0 + ((lane>>4)<<3);
#pragma unroll
  for (int t=0;t<3;++t){
    int col = ng*48 + t*16 + (lane&15);
#pragma unroll
    for (int j=0;j<8;++j){
      int row = rbase + j; int bb = row>>12, ll2 = row&4095;
      float v = acc[t][j];
      size_t base = ((size_t)(bb*LL + ll2))*4 + dir;
      if (col < 12)       dts[base*12 + col] = v;
      else if (col < 76)  Bs [base*64 + (col-12)] = v;
      else if (col < 140) Cs [base*64 + (col-76)] = v;
    }
  }
}

// ---------- chunk-local SSD: Y_diag + chunk states ----------
__global__ void __launch_bounds__(128) k_chunk(const float* __restrict__ dts, const float* __restrict__ Bsg,
                                               const float* __restrict__ Csg, const float* __restrict__ xconv,
                                               const float* __restrict__ A_logs, const float* __restrict__ dt_bias,
                                               float* __restrict__ ys, float* __restrict__ states,
                                               float* __restrict__ expAws, float* __restrict__ cdec){
  int bid = blockIdx.x;                 // ((b*48)+h)*64 + c
  int c  = bid & 63;
  int hb = bid >> 6;
  int h  = hb % 48, b = hb / 48;
  int k  = h / 12, r = h % 12;
  int tid = threadIdx.x, lane = tid & 31, wv = tid >> 5;

  __shared__ __align__(16) __bf16 sC[64*64];
  __shared__ __align__(16) __bf16 sB[64*64];
  __shared__ __align__(16) __bf16 sBdlT[64*64];
  __shared__ __align__(16) __bf16 sXdtT[32*64];
  __shared__ __align__(16) __bf16 sM[64*64];
  __shared__ float sdt[64], sAc[64], sdl[64], sexp[64];
  __shared__ float sAcoef;

  if (tid == 0) sAcoef = -expf(A_logs[k*12+r]);
  if (tid < 64) {
    int l = c*64 + tid;
    float v = dts[(((size_t)(b*LL + l))*4 + k)*12 + r] + dt_bias[k*12+r];
    sdt[tid] = softplus_f(v);
  }
  __syncthreads();
  if (tid == 0){ float acc=0.f; for (int i=0;i<64;++i){ acc += sdt[i]*sAcoef; sAc[i]=acc; } }
  __syncthreads();
  if (tid < 64){ sdl[tid] = expf(sAc[63]-sAc[tid]); sexp[tid] = expf(sAc[tid]); }
  __syncthreads();

  for (int i=tid; i<4096; i+=128){
    int q = i>>6, n = i&63;
    size_t gi = (((size_t)(b*LL + c*64 + q))*4 + k)*64 + n;
    float cv = Csg[gi], bv = Bsg[gi];
    sC[i] = (__bf16)cv;
    sB[i] = (__bf16)bv;
    sBdlT[n*64+q] = (__bf16)(bv * sdl[q]);
  }
  for (int i=tid; i<2048; i+=128){
    int p = i>>6, s = i&63;
    int lm = dirmap(k, c*64 + s);
    float xv = xconv[((size_t)(b*LL + lm))*DI + r*32 + p];
    sXdtT[p*64+s] = (__bf16)(xv * sdt[s]);
  }
  __syncthreads();

  // G = C * B^T, causal decay mask, store as bf16 M
  int q0 = wv*16;
#pragma unroll
  for (int st=0; st<4; ++st){
    f32x8 acc; zero8(acc);
#pragma unroll
    for (int kk=0; kk<64; kk+=32){
      Frag a; load_a(a, sC, 64, q0, kk, lane);
      Frag bfr; load_b(bfr, sB, 64, st*16, kk, lane);
      acc = wmma_bf16(a, bfr, acc);
    }
#pragma unroll
    for (int j=0;j<8;++j){
      int q = q0 + j + ((lane>>4)<<3);
      int s = st*16 + (lane&15);
      float v = acc[j];
      v = (s<=q) ? v*expf(sAc[q]-sAc[s]) : 0.f;
      sM[q*64+s] = (__bf16)v;
    }
  }
  __syncthreads();

  // Y_diag = M * xdt  (64x32)
#pragma unroll
  for (int pt=0; pt<2; ++pt){
    f32x8 acc; zero8(acc);
#pragma unroll
    for (int kk=0; kk<64; kk+=32){
      Frag a; load_a(a, sM, 64, q0, kk, lane);
      Frag bfr; load_b(bfr, sXdtT, 64, pt*16, kk, lane);
      acc = wmma_bf16(a, bfr, acc);
    }
#pragma unroll
    for (int j=0;j<8;++j){
      int q = q0 + j + ((lane>>4)<<3);
      int p = pt*16 + (lane&15);
      ys[(((size_t)(b*LL + c*64 + q))*HH + h)*32 + p] = acc[j];
    }
  }

  // states[p][n] = sum_s xdtT[p][s] * (B[s][n]*dl[s])
  int p0 = (wv>>1)*16;
  size_t sbase = (((size_t)(b*HH + h))*NC + c)*2048;
#pragma unroll
  for (int u=0; u<2; ++u){
    int nt = (wv&1)*2 + u;
    f32x8 acc; zero8(acc);
#pragma unroll
    for (int kk=0; kk<64; kk+=32){
      Frag a; load_a(a, sXdtT, 64, p0, kk, lane);
      Frag bfr; load_b(bfr, sBdlT, 64, nt*16, kk, lane);
      acc = wmma_bf16(a, bfr, acc);
    }
#pragma unroll
    for (int j=0;j<8;++j){
      int p = p0 + j + ((lane>>4)<<3);
      int n = nt*16 + (lane&15);
      states[sbase + p*64 + n] = acc[j];
    }
  }
  if (tid < 64) expAws[(((size_t)(b*HH + h))*NC + c)*64 + tid] = sexp[tid];
  if (tid == 0) cdec[((size_t)(b*HH + h))*NC + c] = expf(sAc[63]);
}

// ---------- inter-chunk recurrence + Y_off ----------
__global__ void __launch_bounds__(128) k_scan(const float* __restrict__ Csg, const float* __restrict__ states,
                                              const float* __restrict__ expAws, const float* __restrict__ cdec,
                                              float* __restrict__ ys){
  int bid = blockIdx.x;     // b*48 + h
  int h = bid % 48, b = bid / 48;
  int k = h / 12;
  int tid = threadIdx.x, lane = tid & 31, wv = tid >> 5;

  __shared__ __align__(16) __bf16 sC[64*64];
  __shared__ __align__(16) __bf16 sSbf[32*64];
  __shared__ float sS[32*64];
  __shared__ float sexp[64];

  for (int i=tid; i<2048; i+=128) sS[i] = 0.f;
  __syncthreads();

  for (int c=0; c<64; ++c){
    for (int i=tid; i<4096; i+=128){
      int q = i>>6, n = i&63;
      sC[i] = (__bf16)Csg[(((size_t)(b*LL + c*64 + q))*4 + k)*64 + n];
    }
    for (int i=tid; i<2048; i+=128) sSbf[i] = (__bf16)sS[i];
    if (tid < 64) sexp[tid] = expAws[(((size_t)(b*HH + h))*NC + c)*64 + tid];
    __syncthreads();

    int q0 = wv*16;
#pragma unroll
    for (int pt=0; pt<2; ++pt){
      f32x8 acc; zero8(acc);
#pragma unroll
      for (int kk=0; kk<64; kk+=32){
        Frag a; load_a(a, sC, 64, q0, kk, lane);
        Frag bfr; load_b(bfr, sSbf, 64, pt*16, kk, lane);
        acc = wmma_bf16(a, bfr, acc);
      }
#pragma unroll
      for (int j=0;j<8;++j){
        int q = q0 + j + ((lane>>4)<<3);
        int p = pt*16 + (lane&15);
        size_t yi = (((size_t)(b*LL + c*64 + q))*HH + h)*32 + p;
        ys[yi] += acc[j] * sexp[q];
      }
    }
    __syncthreads();
    float cd = cdec[((size_t)(b*HH + h))*NC + c];
    size_t sbase = (((size_t)(b*HH + h))*NC + c)*2048;
    for (int i=tid; i<2048; i+=128) sS[i] = cd*sS[i] + states[sbase + i];
    __syncthreads();
  }
}

// ---------- combine directions + D skip ----------
__global__ void k_combine(const float* __restrict__ ys, const float* __restrict__ xconv,
                          const float* __restrict__ Ds, float* __restrict__ yc){
  int idx = blockIdx.x*blockDim.x + threadIdx.x;
  if (idx >= BQ*LL*DI) return;
  int d = idx % DI; int t = idx / DI; int l = t & 4095; int b = t >> 12;
  int r = d >> 5, p = d & 31;
  int l1 = ((l&63)<<6) | (l>>6);
  float v = ys[(((size_t)(b*LL + l      ))*HH + (0*12+r))*32 + p]
          + ys[(((size_t)(b*LL + (4095-l)))*HH + (2*12+r))*32 + p]
          + ys[(((size_t)(b*LL + l1     ))*HH + (1*12+r))*32 + p]
          + ys[(((size_t)(b*LL + (4095-l1)))*HH + (3*12+r))*32 + p];
  float dsum = Ds[(0*12+r)*32+p] + Ds[(1*12+r)*32+p] + Ds[(2*12+r)*32+p] + Ds[(3*12+r)*32+p];
  v += xconv[idx] * dsum;
  yc[idx] = v;
}

// ---------- LayerNorm + gate, emit bf16 ----------
__global__ void __launch_bounds__(128) k_norm(const float* __restrict__ yc, const float* __restrict__ g,
                                              const float* __restrict__ bb, const float* __restrict__ zg,
                                              __bf16* __restrict__ y2){
  int row = blockIdx.x;
  __shared__ float red[128];
  float s=0.f, s2=0.f;
  for (int d=threadIdx.x; d<DI; d+=128){ float v = yc[(size_t)row*DI+d]; s += v; s2 += v*v; }
  red[threadIdx.x]=s; __syncthreads();
  for (int o=64;o>0;o>>=1){ if (threadIdx.x<o) red[threadIdx.x]+=red[threadIdx.x+o]; __syncthreads(); }
  float mean = red[0]/(float)DI; __syncthreads();
  red[threadIdx.x]=s2; __syncthreads();
  for (int o=64;o>0;o>>=1){ if (threadIdx.x<o) red[threadIdx.x]+=red[threadIdx.x+o]; __syncthreads(); }
  float var = red[0]/(float)DI - mean*mean;
  float inv = rsqrtf(var + 1e-5f);
  for (int d=threadIdx.x; d<DI; d+=128){
    float v = (yc[(size_t)row*DI+d]-mean)*inv*g[d] + bb[d];
    y2[(size_t)row*DI+d] = (__bf16)(v * zg[(size_t)row*DI+d]);
  }
}

// ---------- out_proj GEMM: [8192x384] x [192x384]^T ----------
__global__ void __launch_bounds__(256) k_outproj(const __bf16* __restrict__ y2, const __bf16* __restrict__ wo,
                                                 float* __restrict__ out){
  int lane = threadIdx.x & 31;
  int gw = blockIdx.x*8 + (threadIdx.x>>5);
  int mt = gw/3, ng = gw%3;
  int row0 = mt*16, col0 = ng*64;
  f32x8 acc[4];
#pragma unroll
  for (int t=0;t<4;++t) zero8(acc[t]);
#pragma unroll
  for (int kb=0;kb<DI;kb+=32){
    Frag a; load_a(a, y2, DI, row0, kb, lane);
#pragma unroll
    for (int t=0;t<4;++t){ Frag b; load_b(b, wo, DI, col0+t*16, kb, lane); acc[t]=wmma_bf16(a,b,acc[t]); }
  }
  int rbase = row0 + ((lane>>4)<<3);
#pragma unroll
  for (int t=0;t<4;++t){
    int col = col0 + t*16 + (lane&15);
#pragma unroll
    for (int j=0;j<8;++j){
      int row = rbase + j;
      out[(size_t)row*DM + col] = acc[t][j];
    }
  }
}

extern "C" void kernel_launch(void* const* d_in, const int* in_sizes, int n_in,
                              void* d_out, int out_size, void* d_ws, size_t ws_size,
                              hipStream_t stream) {
  (void)in_sizes; (void)n_in; (void)out_size; (void)ws_size;
  const float* x       = (const float*)d_in[0];
  const float* w_in    = (const float*)d_in[1];
  const float* conv_w  = (const float*)d_in[2];
  const float* conv_b  = (const float*)d_in[3];
  const float* w_xp    = (const float*)d_in[4];
  const float* A_logs  = (const float*)d_in[5];
  const float* Ds      = (const float*)d_in[6];
  const float* dt_bias = (const float*)d_in[7];
  const float* ln_g    = (const float*)d_in[8];
  const float* ln_b    = (const float*)d_in[9];
  const float* w_out   = (const float*)d_in[10];
  float* out = (float*)d_out;
  char* ws = (char*)d_ws;

  size_t off = 0;
  auto alloc = [&](size_t bytes){ size_t o = off; off += (bytes + 255) & ~(size_t)255; return o; };
  size_t o_xbf = alloc((size_t)BQ*LL*DM*2);
  size_t o_win = alloc((size_t)768*DM*2);
  size_t o_wxp = alloc((size_t)4*140*DI*2);
  size_t o_wo  = alloc((size_t)DM*DI*2);
  size_t o_xip = alloc((size_t)BQ*LL*DI*4);
  size_t o_zg  = alloc((size_t)BQ*LL*DI*4);
  size_t o_xc  = alloc((size_t)BQ*LL*DI*4);
  size_t o_xcb = alloc((size_t)BQ*LL*DI*2);
  size_t o_dts = alloc((size_t)BQ*LL*HH*4);
  size_t o_Bs  = alloc((size_t)BQ*LL*4*64*4);
  size_t o_Cs  = alloc((size_t)BQ*LL*4*64*4);
  size_t o_ys  = alloc((size_t)BQ*LL*HH*32*4);
  size_t o_st  = alloc((size_t)BQ*HH*NC*32*64*4);
  size_t o_ea  = alloc((size_t)BQ*HH*NC*64*4);
  size_t o_cd  = alloc((size_t)BQ*HH*NC*4);
  size_t o_yc  = alloc((size_t)BQ*LL*DI*4);
  size_t o_y2  = alloc((size_t)BQ*LL*DI*2);

  __bf16* xbf  = (__bf16*)(ws+o_xbf);
  __bf16* winb = (__bf16*)(ws+o_win);
  __bf16* wxpb = (__bf16*)(ws+o_wxp);
  __bf16* wob  = (__bf16*)(ws+o_wo);
  float*  xip  = (float*)(ws+o_xip);
  float*  zg   = (float*)(ws+o_zg);
  float*  xc   = (float*)(ws+o_xc);
  __bf16* xcb  = (__bf16*)(ws+o_xcb);
  float*  dts  = (float*)(ws+o_dts);
  float*  Bs   = (float*)(ws+o_Bs);
  float*  Cs   = (float*)(ws+o_Cs);
  float*  ys   = (float*)(ws+o_ys);
  float*  st   = (float*)(ws+o_st);
  float*  ea   = (float*)(ws+o_ea);
  float*  cd   = (float*)(ws+o_cd);
  float*  yc   = (float*)(ws+o_yc);
  __bf16* y2   = (__bf16*)(ws+o_y2);

  int n;
  n = BQ*LL*DM;     k_cvt_bf16<<<(n+255)/256,256,0,stream>>>(x,     xbf,  n);
  n = 768*DM;       k_cvt_bf16<<<(n+255)/256,256,0,stream>>>(w_in,  winb, n);
  n = 4*140*DI;     k_cvt_bf16<<<(n+255)/256,256,0,stream>>>(w_xp,  wxpb, n);
  n = DM*DI;        k_cvt_bf16<<<(n+255)/256,256,0,stream>>>(w_out, wob,  n);

  k_inproj <<<768,  256, 0, stream>>>(xbf, winb, xip, zg);
  n = BQ*LL*DI;
  k_conv   <<<(n+255)/256, 256, 0, stream>>>(xip, conv_w, conv_b, xc, xcb);
  k_xproj  <<<768,  256, 0, stream>>>(xcb, wxpb, dts, Bs, Cs);
  k_chunk  <<<BQ*HH*NC, 128, 0, stream>>>(dts, Bs, Cs, xc, A_logs, dt_bias, ys, st, ea, cd);
  k_scan   <<<BQ*HH, 128, 0, stream>>>(Cs, st, ea, cd, ys);
  k_combine<<<(n+255)/256, 256, 0, stream>>>(ys, xc, Ds, yc);
  k_norm   <<<BQ*LL, 128, 0, stream>>>(yc, ln_g, ln_b, zg, y2);
  k_outproj<<<192,  256, 0, stream>>>(y2, wob, out);
}
